// AttentionConv2d_27616639714081
// MI455X (gfx1250) — matrix operand order repaired and checked by reading, not compiled, not run on hardware
//
#include <hip/hip_runtime.h>
#include <hip/hip_bf16.h>

// ---------------- CDNA5 WMMA / TDM types ----------------
typedef __attribute__((ext_vector_type(16))) __bf16 v16bf;
typedef __attribute__((ext_vector_type(8)))  float  v8f;
typedef __attribute__((ext_vector_type(4)))  unsigned int v4u;
typedef __attribute__((ext_vector_type(8)))  int v8i;
typedef __attribute__((ext_vector_type(4)))  int v4i;

union BF16Frag { unsigned short u[16]; unsigned int d[8]; v16bf v; };

__device__ __forceinline__ unsigned short f2bf(float f) {
    union { __bf16 b; unsigned short s; } u;
    u.b = (__bf16)f;
    return u.s;
}
__device__ __forceinline__ float bf2f(unsigned short h) {
    return __uint_as_float(((unsigned int)h) << 16);
}
__device__ __forceinline__ unsigned ldsoff(const void* p) {
    return (unsigned)(unsigned long long)p;   // low 32 bits of shared aperture = LDS offset
}

// ---- Tensor Data Mover: 2D tile (tile_w elems x tile_h rows, bf16) -> LDS ----
__device__ __forceinline__ void tdm_load_2d(unsigned lds_addr, const void* gptr,
                                            unsigned tile_w, unsigned tile_h,
                                            unsigned row_stride_elems) {
    unsigned long long ga = (unsigned long long)gptr;
    v4u g0;
    g0[0] = 1u;                                           // count=1 (valid), user mode
    g0[1] = lds_addr;                                     // lds_addr
    g0[2] = (unsigned)(ga & 0xFFFFFFFFu);                 // global_addr[31:0]
    g0[3] = (unsigned)((ga >> 32) & 0x01FFFFFFu) | (2u << 30);  // addr[56:32] | type=2
    v8i g1;
    g1[0] = (int)(1u << 16);                              // data_size=1 (2 bytes)
    g1[1] = (int)((tile_w & 0xFFFFu) << 16);              // tensor_dim0[15:0]
    g1[2] = (int)((tile_w >> 16) | ((tile_h & 0xFFFFu) << 16));  // dim0 hi | dim1 lo
    g1[3] = (int)((tile_h >> 16) | ((tile_w & 0xFFFFu) << 16));  // dim1 hi | tile_dim0
    g1[4] = (int)(tile_h & 0xFFFFu);                      // tile_dim1 (tile_dim2=0 -> 2D)
    g1[5] = (int)row_stride_elems;                        // tensor_dim0_stride[31:0]
    g1[6] = 0;
    g1[7] = 0;
    v4i z = {0, 0, 0, 0};
#if __clang_major__ >= 23
    v8i z8 = {0, 0, 0, 0, 0, 0, 0, 0};
    __builtin_amdgcn_tensor_load_to_lds(g0, g1, z, z, z8, 0);
#else
    __builtin_amdgcn_tensor_load_to_lds(g0, g1, z, z, 0);
#endif
}

// A fragment: 16x32 bf16 row-major (ld even, dword-aligned); dword loads.
__device__ __forceinline__ void loadA_bf16_rm(const unsigned short* base, int ld,
                                              int lane, BF16Frag& A) {
    const unsigned int* row = (const unsigned int*)(base + (lane & 15) * ld);
    int kb2 = (lane & 16) ? 8 : 0;
#pragma unroll
    for (int p = 0; p < 8; ++p) {
        int k2 = kb2 + p;
        A.d[p] = row[k2];
    }
}

// B fragment from K-contiguous storage: B[k][n] = base[n*ldn + k]; dword loads.
__device__ __forceinline__ void loadB_kT(const unsigned short* base, int ldn,
                                         int lane, BF16Frag& B) {
    const unsigned int* col = (const unsigned int*)(base + (lane & 15) * ldn);
    int kb2 = (lane & 16) ? 8 : 0;
#pragma unroll
    for (int p = 0; p < 8; ++p) B.d[p] = col[kb2 + p];
}

// ---------------- 1) conversions / packing ----------------
__global__ void k_cvt(const float* __restrict__ src, unsigned short* __restrict__ dst, int n) {
    int i = blockIdx.x * blockDim.x + threadIdx.x;
    if (i < n) dst[i] = f2bf(src[i]);
}

// x [8,256,1024] fp32 -> Xt [8,1024,256] bf16 (pixel-major: K contiguous)
__global__ void k_cvt_xt(const float* __restrict__ x, unsigned short* __restrict__ Xt) {
    int i = blockIdx.x * blockDim.x + threadIdx.x;     // ci fastest
    if (i >= 8 * 1024 * 256) return;
    int ci = i & 255;
    int l  = (i >> 8) & 1023;
    int b  = i >> 18;
    Xt[i] = f2bf(x[((size_t)b * 256 + ci) * 1024 + l]);
}

__global__ void k_pack_wout(const float* __restrict__ w, unsigned short* __restrict__ dst) {
    int i = blockIdx.x * blockDim.x + threadIdx.x;     // 256*2304
    if (i >= 256 * 2304) return;
    int co = i / 2304, r = i - co * 2304;
    int ci = r / 9, kid = r - ci * 9;
    dst[co * 2304 + kid * 256 + ci] = f2bf(w[i]);      // k-order: kid*256 + ci
}

// ---------------- 2) QKV GEMM: block = 128 M x 16 N, TDM-staged B panel ------
// Panel stored [n][k] (K contiguous): Xt rows are already K-major.
__global__ __launch_bounds__(256) void k_qkv(
    const unsigned short* __restrict__ Xt,  // [8,1024,256] bf16
    const unsigned short* __restrict__ Wq,  // [768,256] bf16
    const float* __restrict__ bqkv,
    unsigned short* __restrict__ Q,         // [64,1024,32]  (scaled)
    unsigned short* __restrict__ K,         // [64,1024,32]
    unsigned short* __restrict__ Vt) {      // [64,32,1024]  (transposed)
    __shared__ unsigned short Bp[16][256];  // 8 KB: 16 pixels x 256 ci (K contiguous)
    const int lane = threadIdx.x & 31;
    const int w    = threadIdx.x >> 5;
    int nt   = blockIdx.x & 63;
    int mblk = (blockIdx.x >> 6) % 6;
    int b    = blockIdx.x / (64 * 6);
    int n0 = nt * 16;
    int m0 = (mblk * 8 + w) * 16;

    if (threadIdx.x < 32) {                 // one wave drives the DMA
        tdm_load_2d(ldsoff(&Bp[0][0]), Xt + (size_t)(b * 1024 + n0) * 256,
                    256, 16, 256);
        __builtin_amdgcn_s_wait_tensorcnt(0);
    }
    __syncthreads();

    const unsigned short* Abase = Wq + m0 * 256;
    int n   = lane & 15;
    int kb2 = (lane & 16) ? 8 : 0;          // dword offset of K base
    BF16Frag Af, Bf;
    v8f C = {};
    for (int kc = 0; kc < 256; kc += 32) {
        loadA_bf16_rm(Abase + kc, 256, lane, Af);
        const unsigned int* brow = (const unsigned int*)&Bp[n][kc];
#pragma unroll
        for (int p = 0; p < 8; ++p) Bf.d[p] = brow[kb2 + p];   // contiguous dwords
        C = __builtin_amdgcn_wmma_f32_16x16x32_bf16(false, Af.v, false, Bf.v,
                                                    (short)0, C, false, false);
    }
    int Mo = (lane & 16) ? 8 : 0;
    const float qscale = 0.17677669529663687f;          // 32^-0.5
#pragma unroll
    for (int e = 0; e < 8; ++e) {
        int c = m0 + e + Mo;
        int l = n0 + n;
        float val = C[e] + bqkv[c];
        if (c < 256) {
            int h = c >> 5, d = c & 31;
            Q[((size_t)(b * 8 + h) * 1024 + l) * 32 + d] = f2bf(val * qscale);
        } else if (c < 512) {
            int c2 = c - 256, h = c2 >> 5, d = c2 & 31;
            K[((size_t)(b * 8 + h) * 1024 + l) * 32 + d] = f2bf(val);
        } else {
            int c2 = c - 512, h = c2 >> 5, d = c2 & 31;
            Vt[((size_t)(b * 8 + h) * 32 + d) * 1024 + l] = f2bf(val);
        }
    }
}

// ---------------- 3) relative-position logit tables ----------------
__global__ void k_rel(const unsigned short* __restrict__ Q,
                      const float* __restrict__ relw,   // [63,32]
                      const float* __restrict__ relh,
                      float* __restrict__ RW, float* __restrict__ RH) {
    int idx = blockIdx.x * blockDim.x + threadIdx.x;    // 64*1024*32
    if (idx >= 64 * 1024 * 32) return;
    int p  = idx & 31;
    int l  = (idx >> 5) & 1023;
    int bh = idx >> 15;
    int y = l >> 5, xx = l & 31;
    const unsigned int* q32 = (const unsigned int*)(Q + ((size_t)bh * 1024 + l) * 32);
    const float* rw = relw + (p - xx + 31) * 32;
    const float* rh = relh + (p - y + 31) * 32;
    float sw = 0.f, sh = 0.f;
#pragma unroll
    for (int d2 = 0; d2 < 16; ++d2) {
        unsigned int qq = q32[d2];
        float q0 = bf2f((unsigned short)(qq & 0xFFFFu));
        float q1 = bf2f((unsigned short)(qq >> 16));
        sw += q0 * rw[2 * d2] + q1 * rw[2 * d2 + 1];
        sh += q0 * rh[2 * d2] + q1 * rh[2 * d2 + 1];
    }
    RW[idx] = sw;
    RH[idx] = sh;
}

// ---------------- 4) fused attention ----------------
__global__ __launch_bounds__(256) void k_attn(
    const unsigned short* __restrict__ Q,
    const unsigned short* __restrict__ K,
    const unsigned short* __restrict__ Vt,  // [64,32,1024]
    const float* __restrict__ RW, const float* __restrict__ RH,
    unsigned short* __restrict__ At /* [8,1024,256] bf16, pixel-major */) {
    __shared__ float          S[16][1024];    // 64 KB scores
    __shared__ unsigned short P[16][1024];    // 32 KB probs (bf16)
    __shared__ float          O[16][32];      // 2 KB output accum

    int bh   = blockIdx.y;
    int q0   = blockIdx.x * 16;
    int lane = threadIdx.x & 31;
    int w    = threadIdx.x >> 5;
    int Mo   = (lane & 16) ? 8 : 0;
    int N    = lane & 15;

    const unsigned short* Qb = Q + ((size_t)bh * 1024 + q0) * 32;
    const unsigned short* Kb = K + (size_t)bh * 1024 * 32;
    const float* RWb = RW + (size_t)bh * 1024 * 32;
    const float* RHb = RH + (size_t)bh * 1024 * 32;

    BF16Frag Af, Bf;
    loadA_bf16_rm(Qb, 32, lane, Af);          // Q tile reused for all col tiles

    // Phase 1: S = Q K^T + rel (64 col tiles, 8 per wave; K==32 -> 1 WMMA each)
    for (int ct = w; ct < 64; ct += 8) {
        int m0 = ct * 16;
        if (ct + 8 < 64)
            __builtin_prefetch(Kb + (size_t)(m0 + 128) * 32, 0, 0);
        loadB_kT(Kb + (size_t)m0 * 32, 32, lane, Bf);
        v8f C = {};
        C = __builtin_amdgcn_wmma_f32_16x16x32_bf16(false, Af.v, false, Bf.v,
                                                    (short)0, C, false, false);
#pragma unroll
        for (int e = 0; e < 8; ++e) {
            int M = e + Mo;
            int l = q0 + M, m = m0 + N;
            S[M][m] = C[e] + RWb[(size_t)l * 32 + (m & 31)] + RHb[(size_t)l * 32 + (m >> 5)];
        }
    }
    __syncthreads();

    // Phase 2: row softmax (2 rows per wave), normalized bf16 P
    for (int rr = 0; rr < 2; ++rr) {
        int r = w * 2 + rr;
        float mx = -3.4e38f;
        for (int j = lane; j < 1024; j += 32) mx = fmaxf(mx, S[r][j]);
        for (int off = 16; off; off >>= 1) mx = fmaxf(mx, __shfl_xor(mx, off, 32));
        float sm = 0.f;
        for (int j = lane; j < 1024; j += 32) {
            float e = __expf(S[r][j] - mx);
            S[r][j] = e;
            sm += e;
        }
        for (int off = 16; off; off >>= 1) sm += __shfl_xor(sm, off, 32);
        float rs = 1.0f / sm;
        for (int j = lane; j < 1024; j += 32) P[r][j] = f2bf(S[r][j] * rs);
    }
    ((float*)O)[threadIdx.x]       = 0.f;
    ((float*)O)[threadIdx.x + 256] = 0.f;
    __syncthreads();

    // Phase 3: O = P (16x1024) * V (1024x32); wave -> (n-tile = w&1, K-quarter)
    int ntile = w & 1;
    int kq    = w >> 1;
    v8f C = {};
    {
        const unsigned int* prow = (const unsigned int*)&P[lane & 15][0];
        int kb2 = (lane & 16) ? 8 : 0;
        const unsigned short* Vbase = Vt + ((size_t)bh * 32 + ntile * 16) * 1024;
        for (int ks = kq * 256; ks < kq * 256 + 256; ks += 32) {
#pragma unroll
            for (int p = 0; p < 8; ++p) {     // A fragment from bf16 P (dword ds)
                int k2 = kb2 + p;
                Af.d[p] = prow[(ks >> 1) + k2];
            }
            loadB_kT(Vbase + ks, 1024, lane, Bf);   // V^T rows: contiguous dwords
            C = __builtin_amdgcn_wmma_f32_16x16x32_bf16(false, Af.v, false, Bf.v,
                                                        (short)0, C, false, false);
        }
    }
#pragma unroll
    for (int e = 0; e < 8; ++e)
        atomicAdd(&O[e + Mo][ntile * 16 + N], C[e]);   // ds_add_f32 K-quarter reduce
    __syncthreads();

    // Write transposed attn map: At[b][pixel][co], co = h*32+d (K-contiguous)
    int b = bh >> 3, h = bh & 7;
#pragma unroll
    for (int i = 0; i < 2; ++i) {
        int idx = threadIdx.x + i * 256;       // 0..511 = 16 rows x 32 dims
        int r = idx >> 5, d = idx & 31;
        At[((size_t)b * 1024 + q0 + r) * 256 + (h * 32 + d)] = f2bf(O[r][d]);
    }
}

// ---------------- 5) output projection: block = 128 M x 16 N, TDM B panel ----
__global__ __launch_bounds__(256) void k_attnproj(
    const unsigned short* __restrict__ At,    // [8,1024,256] bf16 pixel-major
    const unsigned short* __restrict__ Wa,    // [256,256] bf16
    const float* __restrict__ ba,
    float* __restrict__ out /* [8,512,1024] */) {
    __shared__ unsigned short Bp[16][256];    // 8 KB
    const int lane = threadIdx.x & 31;
    const int w    = threadIdx.x >> 5;
    int nt   = blockIdx.x & 63;
    int mblk = (blockIdx.x >> 6) & 1;
    int b    = blockIdx.x >> 7;
    int n0 = nt * 16;
    int m0 = (mblk * 8 + w) * 16;

    if (threadIdx.x < 32) {
        tdm_load_2d(ldsoff(&Bp[0][0]), At + (size_t)(b * 1024 + n0) * 256,
                    256, 16, 256);
        __builtin_amdgcn_s_wait_tensorcnt(0);
    }
    __syncthreads();

    int n   = lane & 15;
    int kb2 = (lane & 16) ? 8 : 0;
    BF16Frag Af, Bf;
    v8f C = {};
    for (int kc = 0; kc < 256; kc += 32) {
        loadA_bf16_rm(Wa + m0 * 256 + kc, 256, lane, Af);
        const unsigned int* brow = (const unsigned int*)&Bp[n][kc];
#pragma unroll
        for (int p = 0; p < 8; ++p) Bf.d[p] = brow[kb2 + p];
        C = __builtin_amdgcn_wmma_f32_16x16x32_bf16(false, Af.v, false, Bf.v,
                                                    (short)0, C, false, false);
    }
    int Mo = (lane & 16) ? 8 : 0;
#pragma unroll
    for (int e = 0; e < 8; ++e) {
        int co = m0 + e + Mo, l = n0 + n;
        out[((size_t)b * 512 + 256 + co) * 1024 + l] = C[e] + ba[co];
    }
}

// ---------------- 6) 3x3 conv: block = 128 co x 16 px, pipelined im2col -----
// Staged tile T[buf][n][k] (K contiguous); one dword gathered per thread/step.
__global__ __launch_bounds__(256) void k_conv3(
    const unsigned short* __restrict__ Xt,    // [8,1024,256] bf16 pixel-major
    const unsigned short* __restrict__ Wc,    // [256,2304] bf16, k = kid*256+ci
    const float* __restrict__ bc,
    float* __restrict__ out) {
    __shared__ unsigned int T[2][16][16];     // 2 x 1 KB im2col tiles (dwords)
    const int lane = threadIdx.x & 31;
    const int w    = threadIdx.x >> 5;
    int nt   = blockIdx.x & 63;
    int mblk = (blockIdx.x >> 6) & 1;
    int b    = blockIdx.x >> 7;
    int n0 = nt * 16;
    int m0 = (mblk * 8 + w) * 16;

    const unsigned short* xb = Xt + (size_t)b * 1024 * 256;
    // staging role for this thread: tile row sn, dword column sk2
    int sn  = threadIdx.x >> 4;               // 0..15 (pixel within tile)
    int sk2 = threadIdx.x & 15;               // 0..15 (dword within 32-K row)
    int spx = n0 + sn;
    int spy = spx >> 5, spxx = spx & 31;

    auto stage = [&](int kc, int buf) {
        int kid = kc >> 8;
        int ky  = kid / 3, kx = kid - ky * 3;
        int sy  = spy + ky - 1, sx = spxx + kx - 1;
        bool ok = ((unsigned)sy < 32u) && ((unsigned)sx < 32u);
        unsigned v = 0;
        if (ok) {
            const unsigned int* src = (const unsigned int*)
                (xb + (size_t)(sy * 32 + sx) * 256 + (kc & 255));
            v = src[sk2];
        }
        T[buf][sn][sk2] = v;
    };

    int n   = lane & 15;
    int kb2 = (lane & 16) ? 8 : 0;            // dword offset of K base within tile
    BF16Frag Af, Bf;
    v8f C = {};
    stage(0, 0);
    for (int s = 0; s < 72; ++s) {
        int kc = s * 32;
        __syncthreads();
        if (s + 1 < 72) stage(kc + 32, (s + 1) & 1);
        loadA_bf16_rm(Wc + (size_t)m0 * 2304 + kc, 2304, lane, Af);
        const unsigned int* brow = &T[s & 1][n][0];
#pragma unroll
        for (int p = 0; p < 8; ++p) Bf.d[p] = brow[kb2 + p];
        C = __builtin_amdgcn_wmma_f32_16x16x32_bf16(false, Af.v, false, Bf.v,
                                                    (short)0, C, false, false);
    }
    int Mo = (lane & 16) ? 8 : 0;
#pragma unroll
    for (int e = 0; e < 8; ++e) {
        int co = m0 + e + Mo, l = n0 + n;
        out[((size_t)b * 512 + co) * 1024 + l] = C[e] + bc[co];
    }
}

// ---------------- host launch ----------------
extern "C" void kernel_launch(void* const* d_in, const int* in_sizes, int n_in,
                              void* d_out, int out_size, void* d_ws, size_t ws_size,
                              hipStream_t stream) {
    const float* x     = (const float*)d_in[0];
    const float* w_qkv = (const float*)d_in[1];
    const float* b_qkv = (const float*)d_in[2];
    const float* w_att = (const float*)d_in[3];
    const float* b_att = (const float*)d_in[4];
    const float* w_out = (const float*)d_in[5];
    const float* b_out = (const float*)d_in[6];
    const float* relw  = (const float*)d_in[7];
    const float* relh  = (const float*)d_in[8];
    float* out = (float*)d_out;                   // [8,512,32,32]

    char* ws = (char*)d_ws;
    size_t off = 0;
    auto carve = [&](size_t bytes) { size_t r = off; off += (bytes + 255) & ~(size_t)255; return r; };
    unsigned short* Wqkv_bf = (unsigned short*)(ws + carve(768 * 256 * 2));
    unsigned short* Watt_bf = (unsigned short*)(ws + carve(256 * 256 * 2));
    unsigned short* Wout_bf = (unsigned short*)(ws + carve(256 * 2304 * 2));
    unsigned short* Xt      = (unsigned short*)(ws + carve((size_t)8 * 1024 * 256 * 2));
    unsigned short* Qbf     = (unsigned short*)(ws + carve((size_t)64 * 1024 * 32 * 2));
    unsigned short* Kbf     = (unsigned short*)(ws + carve((size_t)64 * 1024 * 32 * 2));
    unsigned short* Vtbf    = (unsigned short*)(ws + carve((size_t)64 * 32 * 1024 * 2));
    float*          RW      = (float*)(ws + carve((size_t)64 * 1024 * 32 * 4));
    float*          RH      = (float*)(ws + carve((size_t)64 * 1024 * 32 * 4));
    unsigned short* At      = (unsigned short*)(ws + carve((size_t)8 * 1024 * 256 * 2));
    (void)ws_size; (void)n_in; (void)in_sizes; (void)out_size;

    k_cvt<<<(768 * 256 + 255) / 256, 256, 0, stream>>>(w_qkv, Wqkv_bf, 768 * 256);
    k_cvt<<<(256 * 256 + 255) / 256, 256, 0, stream>>>(w_att, Watt_bf, 256 * 256);
    k_pack_wout<<<(256 * 2304 + 255) / 256, 256, 0, stream>>>(w_out, Wout_bf);
    k_cvt_xt<<<(8 * 1024 * 256) / 256, 256, 0, stream>>>(x, Xt);

    // QKV GEMM: 8 b * 6 mblk(128 rows) * 64 nt
    k_qkv<<<3072, 256, 0, stream>>>(Xt, Wqkv_bf, b_qkv, Qbf, Kbf, Vtbf);

    k_rel<<<(64 * 1024 * 32) / 256, 256, 0, stream>>>(Qbf, relw, relh, RW, RH);

    k_attn<<<dim3(64, 64), 256, 0, stream>>>(Qbf, Kbf, Vtbf, RW, RH, At);

    // projection: 8 b * 2 mblk * 64 nt
    k_attnproj<<<1024, 256, 0, stream>>>(At, Watt_bf, b_att, out);

    // 3x3 conv: 8 b * 2 mblk * 64 nt
    k_conv3<<<1024, 256, 0, stream>>>(Xt, Wout_bf, b_out, out);
}
